// TransformerBlock6_86346022519169
// MI455X (gfx1250) — matrix operand order repaired and checked by reading, not compiled, not run on hardware
//
#include <hip/hip_runtime.h>
#include <hip/hip_bf16.h>

#define BATCH 256
#define SEQ   193
#define TOK   (BATCH * SEQ)   // 49408, divisible by 32
#define DMODEL 64

typedef __attribute__((ext_vector_type(16))) __bf16          v16bf;
typedef __attribute__((ext_vector_type(8)))  float           v8f;
typedef __attribute__((ext_vector_type(8)))  unsigned short  v8u;
typedef __attribute__((ext_vector_type(16))) unsigned short  v16u;

union Frag {
    v16u  u;
    v8u   h[2];
    unsigned short s[16];
    v16bf bf;
};

__device__ __forceinline__ unsigned short f2bf(float f) {
    unsigned int u = __float_as_uint(f);
    unsigned int r = (u + 0x7FFFu + ((u >> 16) & 1u)) >> 16;
    return (unsigned short)r;
}

__device__ __forceinline__ v8f wmma_bf16(v16bf a, v16bf b, v8f c) {
    return __builtin_amdgcn_wmma_f32_16x16x32_bf16(
        /*neg_a=*/false, a, /*neg_b=*/false, b,
        /*c_mod=*/(short)0, c, /*reuse_a=*/false, /*reuse_b=*/false);
}

// A fragment (16x32 bf16, row-major source): lane 0-15 -> rows, two 8-elem K runs.
__device__ __forceinline__ v16bf load_a_lds(const unsigned short* base, int stride,
                                            int row0, int k0, int lane) {
    const int r    = lane & 15;
    const int koff = (lane & 16) ? 8 : 0;
    const unsigned short* p = base + (row0 + r) * stride + k0 + koff;
    Frag f;
    f.h[0] = *(const v8u*)(p);
    f.h[1] = *(const v8u*)(p + 16);
    return f.bf;
}

// B fragment (32x16 bf16, row-major [K][N] source): lane&15 -> N, two 8-elem K runs.
__device__ __forceinline__ v16bf load_b_any(const unsigned short* base, int stride,
                                            int k0, int n0, int lane) {
    const unsigned short* p = base + (k0 + ((lane & 16) ? 8 : 0)) * stride + n0 + (lane & 15);
    Frag f;
#pragma unroll
    for (int e = 0; e < 8; ++e) { f.s[e] = p[e * stride]; f.s[8 + e] = p[(16 + e) * stride]; }
    return f.bf;
}

// C/D fragment store: VGPR r = row (r | r+8 per lane half), lane&15 = column.
__device__ __forceinline__ void store_c(float* base, int stride, int row0, int n0,
                                        int lane, v8f c) {
    const int col = lane & 15;
    const int rb  = (lane & 16) ? 8 : 0;
#pragma unroll
    for (int r = 0; r < 8; ++r)
        base[(row0 + rb + r) * stride + n0 + col] = c[r];
}

// ---------------------------------------------------------------------------
// Kernel 0: convert all weights to bf16 (row-major [K][N], qkv fused to N=192)
// ---------------------------------------------------------------------------
__global__ __launch_bounds__(256) void prep_weights(
    const float* __restrict__ Wq, const float* __restrict__ Wk, const float* __restrict__ Wv,
    const float* __restrict__ WCq, const float* __restrict__ WCk, const float* __restrict__ WCv,
    const float* __restrict__ W1, const float* __restrict__ W2,
    unsigned short* __restrict__ Wqkv, unsigned short* __restrict__ WCqkv,
    unsigned short* __restrict__ W1b, unsigned short* __restrict__ W2b)
{
    const int i = blockIdx.x * 256 + threadIdx.x;
    if (i < 64 * 192) {
        int k = i / 192, n = i % 192;
        const float* W = (n < 64) ? Wq : (n < 128) ? Wk : Wv;
        Wqkv[i] = f2bf(W[k * 64 + (n & 63)]);
    }
    if (i < 1536 * 192) {
        int k = i / 192, n = i % 192;
        const float* W = (n < 64) ? WCq : (n < 128) ? WCk : WCv;
        WCqkv[i] = f2bf(W[k * 64 + (n & 63)]);
    }
    if (i < 64 * 256)  W1b[i] = f2bf(W1[i]);
    if (i < 256 * 64)  W2b[i] = f2bf(W2[i]);
}

// ---------------------------------------------------------------------------
// Kernel 1: gated Q/K/V.  32 tokens per block, 8 waves, WMMA GEMMs:
//   Sqkv = state @ [Wq|Wk|Wv]   (K=64)
//   Cqkv = Ccat  @ [WCq|WCk|WCv](K=1536, 6 chunks of 256 staged bf16 in LDS)
// then q/k/v = LN(S)*LN(C) per 64-wide group.
// ---------------------------------------------------------------------------
__global__ __launch_bounds__(256) void qkv_kernel(
    const float* __restrict__ state,
    const float* __restrict__ C1, const float* __restrict__ C2, const float* __restrict__ C3,
    const float* __restrict__ C4, const float* __restrict__ C5, const float* __restrict__ C6,
    const unsigned short* __restrict__ Wqkv, const unsigned short* __restrict__ WCqkv,
    const float* __restrict__ bq, const float* __restrict__ bk, const float* __restrict__ bv,
    const float* __restrict__ bCq, const float* __restrict__ bCk, const float* __restrict__ bCv,
    const float* __restrict__ gQE, const float* __restrict__ beQE,
    const float* __restrict__ gKE, const float* __restrict__ beKE,
    const float* __restrict__ gVE, const float* __restrict__ beVE,
    const float* __restrict__ gQH, const float* __restrict__ beQH,
    const float* __restrict__ gKH, const float* __restrict__ beKH,
    const float* __restrict__ gVH, const float* __restrict__ beVH,
    float* __restrict__ qb, float* __restrict__ kb, float* __restrict__ vb)
{
    // Overlay: staging (4KB state + 16KB ctx chunk) lives before the barrier,
    // accumulators (2 x 24KB) after it.  48KB total static LDS.
    __shared__ __align__(16) unsigned char smem[49152];
    unsigned short* sA   = (unsigned short*)smem;            // [32][64]  bf16
    unsigned short* cA   = (unsigned short*)(smem + 4096);   // [32][256] bf16
    float*          accS = (float*)smem;                     // [32][192] f32
    float*          accC = (float*)(smem + 24576);           // [32][192] f32

    const int  tid  = threadIdx.x;
    const int  wave = tid >> 5;
    const int  lane = tid & 31;
    const long tok0 = (long)blockIdx.x * 32;

    const float* sp = state + tok0 * 64;
    for (int i = tid; i < 32 * 64; i += 256) sA[i] = f2bf(sp[i]);
    __syncthreads();

    const int mt  = (wave >> 2) * 16;   // M tile row (0 or 16)
    const int ntb = (wave & 3) * 3;     // 3 N tiles of 16 per wave -> 192 cols

    v8f cS[3] = {};
    v8f cC[3] = {};

#pragma unroll
    for (int kk = 0; kk < 64; kk += 32) {
        v16bf a = load_a_lds(sA, 64, mt, kk, lane);
#pragma unroll
        for (int j = 0; j < 3; ++j) {
            v16bf b = load_b_any(Wqkv, 192, kk, (ntb + j) * 16, lane);
            cS[j] = wmma_bf16(a, b, cS[j]);
        }
    }

    const float* Cs[6] = {C1, C2, C3, C4, C5, C6};
    for (int ci = 0; ci < 6; ++ci) {
        __syncthreads();
        const float* cp = Cs[ci] + tok0 * 256;
        for (int i = tid; i < 32 * 256; i += 256) cA[i] = f2bf(cp[i]);
        __syncthreads();
#pragma unroll
        for (int kk = 0; kk < 256; kk += 32) {
            v16bf a = load_a_lds(cA, 256, mt, kk, lane);
#pragma unroll
            for (int j = 0; j < 3; ++j) {
                v16bf b = load_b_any(WCqkv, 192, ci * 256 + kk, (ntb + j) * 16, lane);
                cC[j] = wmma_bf16(a, b, cC[j]);
            }
        }
    }

    __syncthreads();   // retire staging before overlaying accumulators
#pragma unroll
    for (int j = 0; j < 3; ++j) {
        store_c(accS, 192, mt, (ntb + j) * 16, lane, cS[j]);
        store_c(accC, 192, mt, (ntb + j) * 16, lane, cC[j]);
    }
    __syncthreads();

    // LN(S)*LN(C) gating: one thread per (token, projection) pair.
    if (tid < 96) {
        const int t = tid & 31;
        const int p = tid >> 5;   // 0=q 1=k 2=v
        const float* bS = (p == 0) ? bq  : (p == 1) ? bk  : bv;
        const float* bC = (p == 0) ? bCq : (p == 1) ? bCk : bCv;
        const float* gE = (p == 0) ? gQE : (p == 1) ? gKE : gVE;
        const float* bE = (p == 0) ? beQE : (p == 1) ? beKE : beVE;
        const float* gH = (p == 0) ? gQH : (p == 1) ? gKH : gVH;
        const float* bH = (p == 0) ? beQH : (p == 1) ? beKH : beVH;
        float* outp = (p == 0) ? qb : (p == 1) ? kb : vb;

        const float* rS = &accS[t * 192 + p * 64];
        const float* rC = &accC[t * 192 + p * 64];
        float s1 = 0.f, s2 = 0.f, c1 = 0.f, c2 = 0.f;
        for (int d = 0; d < 64; ++d) {
            float x = rS[d] + bS[d]; s1 += x; s2 += x * x;
            float y = rC[d] + bC[d]; c1 += y; c2 += y * y;
        }
        const float mS = s1 * (1.f / 64), vS = s2 * (1.f / 64) - mS * mS;
        const float mC = c1 * (1.f / 64), vC = c2 * (1.f / 64) - mC * mC;
        const float iS = rsqrtf(vS + 1e-5f);
        const float iC = rsqrtf(vC + 1e-5f);
        float* op = outp + (tok0 + t) * 64;
        for (int d = 0; d < 64; ++d) {
            float x = ((rS[d] + bS[d]) - mS) * iS * gE[d] + bE[d];
            float y = ((rC[d] + bC[d]) - mC) * iC * gH[d] + bH[d];
            op[d] = x * y;
        }
    }
}

// ---------------------------------------------------------------------------
// Kernel 2: attention per (b,h). K staged transposed [dk][key] (contiguous
// B-frag runs), V row-major. Flash-style online softmax per 16-row q tile,
// probabilities re-laid out via a 1KB/wave LDS bounce for the P@V WMMAs.
// ---------------------------------------------------------------------------
__global__ __launch_bounds__(256) void attn_kernel(
    const float* __restrict__ qb, const float* __restrict__ kb,
    const float* __restrict__ vb, float* __restrict__ ab)
{
    __shared__ __align__(16) unsigned short KT[32 * 256];     // [dk][key], 16KB
    __shared__ __align__(16) unsigned short Vs[224 * 32];     // [key][dk], 14KB
    __shared__ __align__(16) unsigned short Pb[8 * 16 * 32];  // per-wave P bounce, 8KB

    const int  tid  = threadIdx.x, wave = tid >> 5, lane = tid & 31;
    const int  b    = blockIdx.x >> 1, h = blockIdx.x & 1;
    const long base = ((long)b * SEQ) * 64 + h * 32;

    for (int i = tid; i < 224 * 32; i += 256) {
        int l = i >> 5, d = i & 31;
        float kx = (l < SEQ) ? kb[base + (long)l * 64 + d] : 0.f;
        float vx = (l < SEQ) ? vb[base + (long)l * 64 + d] : 0.f;
        KT[d * 256 + l] = f2bf(kx);
        Vs[i]           = f2bf(vx);
    }
    __syncthreads();

    unsigned short* pw = Pb + wave * (16 * 32);
    const int col = lane & 15;
    const int rb  = (lane & 16) ? 8 : 0;

    for (int qt = wave; qt < 13; qt += 8) {
        // Q A-fragment straight from global with fp32->bf16 conversion
        Frag fq;
        {
            const int r    = lane & 15;
            const int koff = (lane & 16) ? 8 : 0;
            const int l    = qt * 16 + r;
            if (l < SEQ) {
                const float* qp = qb + base + (long)l * 64 + koff;
#pragma unroll
                for (int e = 0; e < 8; ++e) { fq.s[e] = f2bf(qp[e]); fq.s[8 + e] = f2bf(qp[16 + e]); }
            } else {
#pragma unroll
                for (int e = 0; e < 16; ++e) fq.s[e] = 0;
            }
        }
        const v16bf aq = fq.bf;

        float mrow[8], lrow[8];
#pragma unroll
        for (int r = 0; r < 8; ++r) { mrow[r] = -1e30f; lrow[r] = 0.f; }
        v8f o0 = {}, o1 = {};

        for (int c = 0; c < 7; ++c) {           // 32 keys per chunk, 224 padded
            v8f s0 = {}, s1 = {};
            s0 = wmma_bf16(aq, load_b_any(KT, 256, 0, (2 * c) * 16, lane), s0);
            s1 = wmma_bf16(aq, load_b_any(KT, 256, 0, (2 * c + 1) * 16, lane), s1);
            const float scale = 0.17677669529663687f;  // 1/sqrt(32)
            const int  key0 = 2 * c * 16 + col;
            const bool ok0 = key0 < SEQ, ok1 = (key0 + 16) < SEQ;
            float al[8];
#pragma unroll
            for (int r = 0; r < 8; ++r) {
                float x0 = ok0 ? s0[r] * scale : -1e30f;
                float x1 = ok1 ? s1[r] * scale : -1e30f;
                float mc = fmaxf(x0, x1);
#pragma unroll
                for (int off = 1; off < 16; off <<= 1) mc = fmaxf(mc, __shfl_xor(mc, off, 32));
                float mnew  = fmaxf(mrow[r], mc);
                float alpha = __expf(mrow[r] - mnew);
                float p0 = __expf(x0 - mnew);
                float p1 = __expf(x1 - mnew);
                float ls = p0 + p1;
#pragma unroll
                for (int off = 1; off < 16; off <<= 1) ls += __shfl_xor(ls, off, 32);
                lrow[r] = lrow[r] * alpha + ls;
                mrow[r] = mnew;
                al[r]   = alpha;
                s0[r] = p0; s1[r] = p1;
            }
#pragma unroll
            for (int r = 0; r < 8; ++r) {       // rescale + stash probs (C->A relayout)
                o0[r] *= al[r];
                o1[r] *= al[r];
                pw[(rb + r) * 32 + col]      = f2bf(s0[r]);
                pw[(rb + r) * 32 + 16 + col] = f2bf(s1[r]);
            }
            v16bf ap = load_a_lds(pw, 32, 0, 0, lane);
            o0 = wmma_bf16(ap, load_b_any(Vs, 32, c * 32, 0, lane), o0);
            o1 = wmma_bf16(ap, load_b_any(Vs, 32, c * 32, 16, lane), o1);
        }

#pragma unroll
        for (int r = 0; r < 8; ++r) {
            int l = qt * 16 + rb + r;
            if (l < SEQ) {
                float inv = 1.f / lrow[r];
                ab[base + (long)l * 64 + col]      = o0[r] * inv;
                ab[base + (long)l * 64 + 16 + col] = o1[r] * inv;
            }
        }
    }
}

// ---------------------------------------------------------------------------
// Kernel 3: Z1 = state+attn; Z2 = LN(Z1); GEMM1+GELU(erf); GEMM2; out = Z1+Z4
// ---------------------------------------------------------------------------
__global__ __launch_bounds__(256) void ffn_kernel(
    const float* __restrict__ state, const float* __restrict__ ab,
    const float* __restrict__ g1, const float* __restrict__ be1,
    const unsigned short* __restrict__ W1b, const float* __restrict__ bl1,
    const unsigned short* __restrict__ W2b, const float* __restrict__ bl2,
    float* __restrict__ out)
{
    __shared__ float                          z1[32 * 64];    // 8KB
    __shared__ __align__(16) unsigned short   z2[32 * 64];    // 4KB bf16
    __shared__ __align__(16) unsigned short   h3[32 * 256];   // 16KB bf16

    const int  tid  = threadIdx.x, wave = tid >> 5, lane = tid & 31;
    const long tok0 = (long)blockIdx.x * 32;

    for (int i = tid; i < 2048; i += 256)
        z1[i] = state[tok0 * 64 + i] + ab[tok0 * 64 + i];
    __syncthreads();

    if (tid < 32) {
        const float* r = &z1[tid * 64];
        float s1 = 0.f, s2 = 0.f;
        for (int d = 0; d < 64; ++d) { s1 += r[d]; s2 += r[d] * r[d]; }
        const float m  = s1 * (1.f / 64);
        const float v  = s2 * (1.f / 64) - m * m;
        const float rs = rsqrtf(v + 1e-5f);
        for (int d = 0; d < 64; ++d)
            z2[tid * 64 + d] = f2bf((r[d] - m) * rs * g1[d] + be1[d]);
    }
    __syncthreads();

    const int mt  = (wave >> 2) * 16;
    const int col = lane & 15;
    const int rb  = (lane & 16) ? 8 : 0;

    {   // GEMM1: [32x64] @ [64x256], 4 N-tiles per wave
        const int ntb = (wave & 3) * 4;
        v8f hc[4] = {};
#pragma unroll
        for (int kk = 0; kk < 64; kk += 32) {
            v16bf a = load_a_lds(z2, 64, mt, kk, lane);
#pragma unroll
            for (int j = 0; j < 4; ++j)
                hc[j] = wmma_bf16(a, load_b_any(W1b, 256, kk, (ntb + j) * 16, lane), hc[j]);
        }
#pragma unroll
        for (int j = 0; j < 4; ++j) {
            const int   n0   = (ntb + j) * 16;
            const float bias = bl1[n0 + col];
#pragma unroll
            for (int r = 0; r < 8; ++r) {
                float x = hc[j][r] + bias;
                float g = 0.5f * x * (1.f + erff(x * 0.70710678118654752f));
                h3[(mt + rb + r) * 256 + n0 + col] = f2bf(g);
            }
        }
    }
    __syncthreads();

    {   // GEMM2: [32x256] @ [256x64], one 16x16 tile per wave
        const int nt = wave & 3;
        v8f oc = {};
#pragma unroll
        for (int kk = 0; kk < 256; kk += 32) {
            v16bf a = load_a_lds(h3, 256, mt, kk, lane);
            oc = wmma_bf16(a, load_b_any(W2b, 64, kk, nt * 16, lane), oc);
        }
        const float bias = bl2[nt * 16 + col];
#pragma unroll
        for (int r = 0; r < 8; ++r) {
            const int row = mt + rb + r;
            out[tok0 * 64 + row * 64 + nt * 16 + col] =
                z1[row * 64 + nt * 16 + col] + oc[r] + bias;
        }
    }
}

// ---------------------------------------------------------------------------
extern "C" void kernel_launch(void* const* d_in, const int* in_sizes, int n_in,
                              void* d_out, int out_size, void* d_ws, size_t ws_size,
                              hipStream_t stream) {
    const float* state = (const float*)d_in[0];
    const float* C1 = (const float*)d_in[1];
    const float* C2 = (const float*)d_in[2];
    const float* C3 = (const float*)d_in[3];
    const float* C4 = (const float*)d_in[4];
    const float* C5 = (const float*)d_in[5];
    const float* C6 = (const float*)d_in[6];
    const float* Wq = (const float*)d_in[7];
    const float* Wk = (const float*)d_in[8];
    const float* Wv = (const float*)d_in[9];
    const float* bq = (const float*)d_in[10];
    const float* bk = (const float*)d_in[11];
    const float* bv = (const float*)d_in[12];
    const float* WCq = (const float*)d_in[13];
    const float* WCk = (const float*)d_in[14];
    const float* WCv = (const float*)d_in[15];
    const float* bCq = (const float*)d_in[16];
    const float* bCk = (const float*)d_in[17];
    const float* bCv = (const float*)d_in[18];
    const float* gQE = (const float*)d_in[19];
    const float* beQE = (const float*)d_in[20];
    const float* gKE = (const float*)d_in[21];
    const float* beKE = (const float*)d_in[22];
    const float* gVE = (const float*)d_in[23];
    const float* beVE = (const float*)d_in[24];
    const float* gQH = (const float*)d_in[25];
    const float* beQH = (const float*)d_in[26];
    const float* gKH = (const float*)d_in[27];
    const float* beKH = (const float*)d_in[28];
    const float* gVH = (const float*)d_in[29];
    const float* beVH = (const float*)d_in[30];
    const float* g1 = (const float*)d_in[31];
    const float* be1 = (const float*)d_in[32];
    const float* W1 = (const float*)d_in[33];
    const float* bl1 = (const float*)d_in[34];
    const float* W2 = (const float*)d_in[35];
    const float* bl2 = (const float*)d_in[36];
    float* out = (float*)d_out;

    // Workspace layout (~51.3 MB)
    float* qb = (float*)d_ws;
    float* kb = qb + (size_t)TOK * 64;
    float* vb = kb + (size_t)TOK * 64;
    float* ab = vb + (size_t)TOK * 64;
    unsigned short* Wqkv  = (unsigned short*)(ab + (size_t)TOK * 64);
    unsigned short* WCqkv = Wqkv + 64 * 192;
    unsigned short* W1b   = WCqkv + 1536 * 192;
    unsigned short* W2b   = W1b + 64 * 256;

    prep_weights<<<1152, 256, 0, stream>>>(Wq, Wk, Wv, WCq, WCk, WCv, W1, W2,
                                           Wqkv, WCqkv, W1b, W2b);

    qkv_kernel<<<TOK / 32, 256, 0, stream>>>(
        state, C1, C2, C3, C4, C5, C6, Wqkv, WCqkv,
        bq, bk, bv, bCq, bCk, bCv,
        gQE, beQE, gKE, beKE, gVE, beVE,
        gQH, beQH, gKH, beKH, gVH, beVH,
        qb, kb, vb);

    attn_kernel<<<BATCH * 2, 256, 0, stream>>>(qb, kb, vb, ab);

    ffn_kernel<<<TOK / 32, 256, 0, stream>>>(state, ab, g1, be1, W1b, bl1, W2b, bl2, out);
}
